// PELICANBlock_65163243815404
// MI455X (gfx1250) — matrix operand order repaired and checked by reading, not compiled, not run on hardware
//
#include <hip/hip_runtime.h>

// ---------------------------------------------------------------------------
// PELICAN block for MI455X (gfx1250, wave32, WMMA).
// Dense reformulation: t = LN(gelu(x @ W^T + b)) viewed as (B,N,N,C);
// out = gelu(P0 t + P1 t^T + i-terms + j-terms + batch-terms + eye-terms).
// Big GEMMs (3x 131072x128x128) via V_WMMA_F32_16X16X32_BF16, f32 accum.
// ---------------------------------------------------------------------------

#define NB 32
#define NN 64
#define NC 128
#define NE (NB * NN * NN) // 131072

typedef __bf16 v16bf __attribute__((ext_vector_type(16)));
typedef float  v8f   __attribute__((ext_vector_type(8)));

struct __align__(16) U4 { unsigned int a, b, c, d; };
struct U4x2 { U4 lo, hi; };

static __device__ __forceinline__ v16bf make_frag(U4 lo, U4 hi) {
  U4x2 t; t.lo = lo; t.hi = hi;
  return __builtin_bit_cast(v16bf, t);
}

static __device__ __forceinline__ v8f wmma_bf16(v16bf a, v16bf b, v8f c) {
  // D = A(16x32 bf16) x B(32x16 bf16) + C(16x16 f32)
  return __builtin_amdgcn_wmma_f32_16x16x32_bf16(false, a, false, b,
                                                 (short)0, c, false, false);
}

static __device__ __forceinline__ unsigned short f2bf(float f) {
  unsigned int u = __builtin_bit_cast(unsigned int, f);
  u = (u + 0x7FFFu + ((u >> 16) & 1u)) >> 16; // RNE
  return (unsigned short)u;
}
static __device__ __forceinline__ unsigned int packbf2(float a, float b) {
  return (unsigned int)f2bf(a) | ((unsigned int)f2bf(b) << 16);
}
static __device__ __forceinline__ float bf2f(unsigned short u) {
  unsigned int x = ((unsigned int)u) << 16;
  return __builtin_bit_cast(float, x);
}
static __device__ __forceinline__ float gelu_exact(float x) {
  return 0.5f * x * (1.0f + erff(x * 0.70710678118654752440f));
}

// ---------------------------------------------------------------------------
// Kernel P: build the 15 combined maps P_m[o,c] = c00[o,m]*c10[o,c]
//           + c01[c,m]*c11[o,c]; bf16 copies of P0/P1 for the main WMMA GEMMs.
// ---------------------------------------------------------------------------
__global__ __launch_bounds__(128) void k_prep(
    const float* __restrict__ c00, const float* __restrict__ c01,
    const float* __restrict__ c10, const float* __restrict__ c11,
    float* __restrict__ Pm, unsigned short* __restrict__ P0bf,
    unsigned short* __restrict__ P1bf) {
  int o = blockIdx.x, c = threadIdx.x;
  float v10 = c10[o * NC + c], v11 = c11[o * NC + c];
#pragma unroll
  for (int m = 0; m < 15; ++m) {
    float v = c00[o * 15 + m] * v10 + c01[c * 15 + m] * v11;
    Pm[m * NC * NC + o * NC + c] = v;
    if (m == 0) P0bf[o * NC + c] = f2bf(v);
    if (m == 1) P1bf[o * NC + c] = f2bf(v);
  }
}

// ---------------------------------------------------------------------------
// Kernel A: per (b,i) tile of 64 edges: h = gelu(x@W^T+b), LayerNorm,
// write t as bf16, row-mean (over j, tile-local) and diag row.
// 256 threads = 8 waves; wave w: M-tile (w&3), N-half (w>>2). 16 WMMAs/wave.
// ---------------------------------------------------------------------------
__global__ __launch_bounds__(256) void k_embed(
    const float* __restrict__ x, const float* __restrict__ W,
    const float* __restrict__ bias, const float* __restrict__ lng,
    const float* __restrict__ lnb, unsigned int* __restrict__ hb,
    float* __restrict__ rowm, float* __restrict__ diag) {
  __shared__ __align__(16) unsigned int sX[64 * 64]; // 64 rows x 128 bf16
  __shared__ __align__(16) unsigned int sW[128 * 64];
  __shared__ __align__(16) float sG[64 * 128];
  __shared__ float sPS[256], sPS2[256];

  const int t = threadIdx.x, blk = blockIdx.x, iloc = blk & 63;

  // stage x tile + W as bf16 into LDS (row-major, K-pairs per dword)
  const float2* x2 = (const float2*)(x + (size_t)blk * (64 * NC));
#pragma unroll
  for (int it = 0; it < 16; ++it) {
    int pi = t + it * 256;
    float2 v = x2[pi];
    sX[pi] = packbf2(v.x, v.y);
  }
  const float2* w2 = (const float2*)W;
#pragma unroll
  for (int it = 0; it < 32; ++it) {
    int pi = t + it * 256;
    float2 v = w2[pi];
    sW[pi] = packbf2(v.x, v.y);
  }
  __syncthreads();

  const int wave = t >> 5, lane = t & 31, mrow = lane & 15, half = lane >> 4;
  const int m0 = (wave & 3) * 16, nh = wave >> 2;
  const U4* a4 = (const U4*)sX;
  const U4* b4 = (const U4*)sW;
  v8f acc[4] = {};
#pragma unroll
  for (int k = 0; k < 4; ++k) {
    int ai = (m0 + mrow) * 16 + k * 4 + half; // A: K pairs, half-wave K+8
    v16bf af = make_frag(a4[ai], a4[ai + 2]);
#pragma unroll
    for (int nt = 0; nt < 4; ++nt) {
      int n = (nh * 4 + nt) * 16 + mrow;
      int bi = n * 16 + k * 4 + half;         // B: half-wave K+8
      v16bf bf = make_frag(b4[bi], b4[bi + 2]);
      acc[nt] = wmma_bf16(af, bf, acc[nt]);
    }
  }
  // bias + exact GELU -> f32 scratch tile
#pragma unroll
  for (int nt = 0; nt < 4; ++nt) {
    int n = (nh * 4 + nt) * 16 + mrow;
    float bb = bias[n];
#pragma unroll
    for (int r = 0; r < 8; ++r) {
      int m = m0 + r + 8 * half;
      sG[m * NC + n] = gelu_exact(acc[nt][r] + bb);
    }
  }
  __syncthreads();

  // LayerNorm over C: 4 threads per row
  const int row = t >> 2, q = t & 3;
  float s = 0.f, s2 = 0.f;
#pragma unroll
  for (int u = 0; u < 32; ++u) {
    float v = sG[row * NC + q * 32 + u];
    s += v; s2 += v * v;
  }
  sPS[row * 4 + q] = s; sPS2[row * 4 + q] = s2;
  __syncthreads();
  float mu = (sPS[row * 4] + sPS[row * 4 + 1] + sPS[row * 4 + 2] + sPS[row * 4 + 3]) * (1.f / 128.f);
  float ex2 = (sPS2[row * 4] + sPS2[row * 4 + 1] + sPS2[row * 4 + 2] + sPS2[row * 4 + 3]) * (1.f / 128.f);
  float rstd = rsqrtf(ex2 - mu * mu + 1e-5f);
  unsigned int* hrow = hb + (size_t)blk * (64 * NC / 2) + row * (NC / 2);
#pragma unroll
  for (int u = 0; u < 16; ++u) {
    int c = q * 32 + 2 * u;
    float v0 = (sG[row * NC + c] - mu) * rstd * lng[c] + lnb[c];
    float v1 = (sG[row * NC + c + 1] - mu) * rstd * lng[c + 1] + lnb[c + 1];
    hrow[q * 16 + u] = packbf2(v0, v1);
    sG[row * NC + c] = v0; sG[row * NC + c + 1] = v1;
    if (row == iloc) { diag[(size_t)blk * NC + c] = v0; diag[(size_t)blk * NC + c + 1] = v1; }
  }
  __syncthreads();
  if (t < NC) {
    float ss = 0.f;
    for (int j = 0; j < 64; ++j) ss += sG[j * NC + t];
    rowm[(size_t)blk * NC + t] = ss * (1.f / 64.f);
  }
}

// ---------------------------------------------------------------------------
// Kernel B: col[b,j,c] = mean_i t[b,i,j,c]
// ---------------------------------------------------------------------------
__global__ __launch_bounds__(128) void k_col(const unsigned short* __restrict__ hb,
                                             float* __restrict__ colm) {
  int blk = blockIdx.x, b = blk >> 6, j = blk & 63, c = threadIdx.x;
  float s = 0.f;
  for (int i = 0; i < 64; ++i)
    s += bf2f(hb[((size_t)(b * 64 + i) * 64 + j) * NC + c]);
  colm[(size_t)blk * NC + c] = s * (1.f / 64.f);
}

// ---------------------------------------------------------------------------
// Kernel B2: trace/tot per batch + Cb = P8 trace + P9 tot
// ---------------------------------------------------------------------------
__global__ __launch_bounds__(128) void k_batch(
    const float* __restrict__ diag, const float* __restrict__ rowm,
    const float* __restrict__ Pm, float* __restrict__ trace,
    float* __restrict__ tot, float* __restrict__ Cb) {
  __shared__ float str[NC], sto[NC];
  int b = blockIdx.x, c = threadIdx.x;
  float tr = 0.f, to = 0.f;
  for (int i = 0; i < 64; ++i) {
    tr += diag[(size_t)(b * 64 + i) * NC + c];
    to += rowm[(size_t)(b * 64 + i) * NC + c];
  }
  tr *= (1.f / 64.f); to *= (1.f / 64.f);
  trace[b * NC + c] = tr; tot[b * NC + c] = to;
  str[c] = tr; sto[c] = to;
  __syncthreads();
  int o = threadIdx.x;
  const float* P8 = Pm + 8 * NC * NC + o * NC;
  const float* P9 = Pm + 9 * NC * NC + o * NC;
  float acc = 0.f;
  for (int k = 0; k < NC; ++k) acc += P8[k] * str[k] + P9[k] * sto[k];
  Cb[b * NC + o] = acc;
}

// ---------------------------------------------------------------------------
// Kernel C: per (b,idx) node terms:
//  A = P2 d + P4 r + P6 c (consumed at i), Bt = P3 d + P5 r + P7 c (at j),
//  Eb = P10 d + P11 r + P12 c + P13 tr + P14 to (at i==j).
// ---------------------------------------------------------------------------
__global__ __launch_bounds__(128) void k_node(
    const float* __restrict__ diag, const float* __restrict__ rowm,
    const float* __restrict__ colm, const float* __restrict__ trace,
    const float* __restrict__ tot, const float* __restrict__ Pm,
    float* __restrict__ Aterm, float* __restrict__ Bterm,
    float* __restrict__ Eterm) {
  __shared__ float sd[NC], sr[NC], sc[NC], st[NC], so[NC];
  int blk = blockIdx.x, b = blk >> 6, t = threadIdx.x;
  sd[t] = diag[(size_t)blk * NC + t];
  sr[t] = rowm[(size_t)blk * NC + t];
  sc[t] = colm[(size_t)blk * NC + t];
  st[t] = trace[b * NC + t];
  so[t] = tot[b * NC + t];
  __syncthreads();
  int o = t;
  const float* P2  = Pm + 2  * NC * NC + o * NC;
  const float* P3  = Pm + 3  * NC * NC + o * NC;
  const float* P4  = Pm + 4  * NC * NC + o * NC;
  const float* P5  = Pm + 5  * NC * NC + o * NC;
  const float* P6  = Pm + 6  * NC * NC + o * NC;
  const float* P7  = Pm + 7  * NC * NC + o * NC;
  const float* P10 = Pm + 10 * NC * NC + o * NC;
  const float* P11 = Pm + 11 * NC * NC + o * NC;
  const float* P12 = Pm + 12 * NC * NC + o * NC;
  const float* P13 = Pm + 13 * NC * NC + o * NC;
  const float* P14 = Pm + 14 * NC * NC + o * NC;
  float aA = 0.f, aB = 0.f, aE = 0.f;
  for (int k = 0; k < NC; ++k) {
    float d = sd[k], r = sr[k], c = sc[k], tr = st[k], to = so[k];
    aA += P2[k] * d + P4[k] * r + P6[k] * c;
    aB += P3[k] * d + P5[k] * r + P7[k] * c;
    aE += P10[k] * d + P11[k] * r + P12[k] * c + P13[k] * tr + P14[k] * to;
  }
  Aterm[(size_t)blk * NC + o] = aA;
  Bterm[(size_t)blk * NC + o] = aB;
  Eterm[(size_t)blk * NC + o] = aE;
}

// ---------------------------------------------------------------------------
// Kernel D (main): block = (b, itile, jtile) 16x16 edge tile, 512 thr = 16
// waves; wave w owns i = itile*16+w, j = j0..j0+15. Stages t[b,i,*] and
// t[b,*,i] rows to LDS, runs 64 bf16 WMMAs (P0 t + P1 t^T), fuses broadcast
// terms + eye + exact GELU into the f32 store.
// ---------------------------------------------------------------------------
__global__ __launch_bounds__(512) void k_main(
    const unsigned short* __restrict__ hb, const unsigned short* __restrict__ P0bf,
    const unsigned short* __restrict__ P1bf, const float* __restrict__ Aterm,
    const float* __restrict__ Bterm, const float* __restrict__ Eterm,
    const float* __restrict__ Cb, float* __restrict__ out) {
  __shared__ __align__(16) unsigned int sAB[32768]; // 128 KB: A1 | A2
  const int tid = threadIdx.x, wave = tid >> 5, lane = tid & 31;
  const int blk = blockIdx.x, b = blk >> 4, it_ = (blk >> 2) & 3, jt = blk & 3;
  const int i = it_ * 16 + wave, j0 = jt * 16;

  U4* s4 = (U4*)sAB;
  const U4* h4 = (const U4*)hb; // 16 U4 per 128-ch bf16 row
  {
    const int rr = lane >> 1, hf = lane & 1;
    size_t e1 = ((size_t)(b * 64 + i)) * 64 + (j0 + rr);
    size_t e2 = ((size_t)(b * 64 + j0 + rr)) * 64 + i;
    U4* d1 = s4 + (wave * 16 + rr) * 16 + hf * 8;
    U4* d2 = d1 + 4096;
    const U4* p1 = h4 + e1 * 16 + hf * 8;
    const U4* p2 = h4 + e2 * 16 + hf * 8;
#pragma unroll
    for (int q = 0; q < 8; ++q) { d1[q] = p1[q]; d2[q] = p2[q]; }
  }
  __syncthreads();

  const int mrow = lane & 15, half = lane >> 4;
  v8f acc[8] = {};
#pragma unroll
  for (int g = 0; g < 2; ++g) {
    const U4* A4 = s4 + g * 4096 + wave * 16 * 16;
    const U4* Pb = (const U4*)(g == 0 ? P0bf : P1bf);
#pragma unroll
    for (int k = 0; k < 4; ++k) {
      int ai = mrow * 16 + k * 4 + half;
      v16bf af = make_frag(A4[ai], A4[ai + 2]);
#pragma unroll
      for (int nt = 0; nt < 8; ++nt) {
        int n = nt * 16 + mrow;
        int bi = n * 16 + k * 4 + half;
        v16bf bf = make_frag(Pb[bi], Pb[bi + 2]);
        acc[nt] = wmma_bf16(af, bf, acc[nt]);
      }
    }
  }

  const float* At = Aterm + (size_t)(b * 64 + i) * NC;
  const float* Et = Eterm + (size_t)(b * 64 + i) * NC;
  const float* Cp = Cb + b * NC;
  float* op = out + (size_t)(b * 64 + i) * 64 * NC;
#pragma unroll
  for (int nt = 0; nt < 8; ++nt) {
    int n = nt * 16 + mrow;
    float aI = At[n] + Cp[n];
#pragma unroll
    for (int r = 0; r < 8; ++r) {
      int j = j0 + r + 8 * half;
      float v = acc[nt][r] + aI + Bterm[(size_t)(b * 64 + j) * NC + n];
      if (j == i) v += Et[n];
      op[(size_t)j * NC + n] = gelu_exact(v);
    }
  }
}

// ---------------------------------------------------------------------------
extern "C" void kernel_launch(void* const* d_in, const int* in_sizes, int n_in,
                              void* d_out, int out_size, void* d_ws, size_t ws_size,
                              hipStream_t stream) {
  const float* x    = (const float*)d_in[0];
  // d_in[1]=edge_index, d_in[2]=batch: fully regular -> unused
  const float* W    = (const float*)d_in[3];
  const float* b_in = (const float*)d_in[4];
  const float* lng  = (const float*)d_in[5];
  const float* lnb  = (const float*)d_in[6];
  const float* c00  = (const float*)d_in[7];
  const float* c01  = (const float*)d_in[8];
  const float* c10  = (const float*)d_in[9];
  const float* c11  = (const float*)d_in[10];

  char* w = (char*)d_ws;
  unsigned int*   hb    = (unsigned int*)w;           w += (size_t)NE * NC * 2;      // t as bf16, 32 MB
  float*          rowm  = (float*)w;                  w += (size_t)NB * NN * NC * 4; // 1 MB
  float*          colm  = (float*)w;                  w += (size_t)NB * NN * NC * 4;
  float*          diag  = (float*)w;                  w += (size_t)NB * NN * NC * 4;
  float*          trace = (float*)w;                  w += (size_t)NB * NC * 4;
  float*          tot   = (float*)w;                  w += (size_t)NB * NC * 4;
  float*          Cb    = (float*)w;                  w += (size_t)NB * NC * 4;
  float*          Aterm = (float*)w;                  w += (size_t)NB * NN * NC * 4;
  float*          Bterm = (float*)w;                  w += (size_t)NB * NN * NC * 4;
  float*          Eterm = (float*)w;                  w += (size_t)NB * NN * NC * 4;
  float*          Pm    = (float*)w;                  w += (size_t)15 * NC * NC * 4; // ~0.94 MB
  unsigned short* P0bf  = (unsigned short*)w;         w += (size_t)NC * NC * 2;
  unsigned short* P1bf  = (unsigned short*)w;         w += (size_t)NC * NC * 2;
  (void)ws_size; (void)in_sizes; (void)n_in; (void)out_size;

  k_prep <<<128, 128, 0, stream>>>(c00, c01, c10, c11, Pm, P0bf, P1bf);
  k_embed<<<NB * NN, 256, 0, stream>>>(x, W, b_in, lng, lnb, hb, rowm, diag);
  k_col  <<<NB * NN, 128, 0, stream>>>((const unsigned short*)hb, colm);
  k_batch<<<NB, 128, 0, stream>>>(diag, rowm, Pm, trace, tot, Cb);
  k_node <<<NB * NN, 128, 0, stream>>>(diag, rowm, colm, trace, tot, Pm,
                                       Aterm, Bterm, Eterm);
  k_main <<<512, 512, 0, stream>>>((const unsigned short*)hb, P0bf, P1bf,
                                   Aterm, Bterm, Eterm, Cb, (float*)d_out);
}
